// PatchCoherentLoss_37168646980355
// MI455X (gfx1250) — matrix operand order repaired and checked by reading, not compiled, not run on hardware
//
#include <hip/hip_runtime.h>
#include <stdint.h>

// ---------------------------------------------------------------------------
// PatchCoherentLoss for MI455X (gfx1250, wave32, WMMA).
//
// dist = (|xp|^2 + |yp|^2 - 2 xp@yp^T)/504 over 8186x8186x504.
// GEMM done with V_WMMA_F32_16X16X32_F16 using f16 hi/lo split precision
// (Ah*Bh + Ah*Bl + Al*Bh, f32 accumulate) ~= fp32 accuracy at f16-WMMA rate.
// Two GEMM passes (colmin, then normalized row-argmin) avoid a 268MB dist
// buffer; the 32MB patch matrices stay resident in the 192MB L2.
// Staging uses GLOBAL_LOAD_ASYNC_TO_LDS_B128 (ASYNCcnt) with double-buffered
// LDS when the toolchain exposes the builtin; sync ds_store path otherwise.
// ---------------------------------------------------------------------------

typedef __attribute__((ext_vector_type(16))) _Float16 v16h;
typedef __attribute__((ext_vector_type(8)))  float    v8f;

#define C_CH     72
#define T_LEN    8192
#define PATCH    7
#define N1       8186          // T - PATCH + 1 (valid patches)
#define NPAD     8192          // padded to 128-multiple
#define KD       504           // C_CH * PATCH
#define KPAD     512
#define INV_D    (1.0f / 504.0f)
#define ALPHA_C  0.01f
#define LDS_PITCH 40           // halfs per LDS row (16B aligned, conflict-lean)

#if __has_builtin(__builtin_amdgcn_global_load_async_to_lds_b128) && \
    __has_builtin(__builtin_amdgcn_s_wait_asynccnt)
#define USE_ASYNC_LDS 1
#else
#define USE_ASYNC_LDS 0
#endif

#if USE_ASYNC_LDS
// Builtin expects pointers to GCC-vector int4 (16B payload):
//   arg0: AS(1) global source, arg1: AS(3) LDS destination, imm offset, imm cpol.
typedef int v4i_vs __attribute__((vector_size(4 * sizeof(int))));
typedef __attribute__((address_space(1))) v4i_vs GV4;
typedef __attribute__((address_space(3))) v4i_vs LV4;
// 16B per lane, global -> LDS, no VGPR round trip; tracked by ASYNCcnt.
__device__ inline void async_cp16(const void* g, const void* l) {
  __builtin_amdgcn_global_load_async_to_lds_b128(
      (GV4*)(uintptr_t)g,
      (LV4*)(unsigned)(uintptr_t)l,   // low 32 bits of generic = LDS offset
      0, 0);
}
#endif

// ---------------------------------------------------------------- init ----
__global__ void init_kernel(unsigned* colmin_bits, unsigned long long* best) {
  int j = blockIdx.x * 256 + threadIdx.x;
  if (j < NPAD) { colmin_bits[j] = 0xFFFFFFFFu; best[j] = ~0ull; }
}

// ------------------------------------------------- build patch matrices ----
// Xp[i][k] = X[c, i+p], k = c*7+p; padded with zeros; split into f16 hi+lo.
__global__ void prep_patches(const float* __restrict__ X, const float* __restrict__ Y,
                             _Float16* XpH, _Float16* XpL,
                             _Float16* YpH, _Float16* YpL) {
  int idx = blockIdx.x * 256 + threadIdx.x;
  if (idx >= NPAD * KPAD) return;
  int i = idx >> 9;          // / KPAD
  int k = idx & (KPAD - 1);
  float x = 0.0f, y = 0.0f;
  if (i < N1 && k < KD) {
    int c = k / PATCH, p = k - c * PATCH;
    x = X[c * T_LEN + i + p];
    y = Y[c * T_LEN + i + p];
  }
  _Float16 xh = (_Float16)x; _Float16 xl = (_Float16)(x - (float)xh);
  _Float16 yh = (_Float16)y; _Float16 yl = (_Float16)(y - (float)yh);
  XpH[idx] = xh; XpL[idx] = xl; YpH[idx] = yh; YpL[idx] = yl;
}

// ------------------------------------------------------- patch norms ------
__global__ void norms_kernel(const float* __restrict__ X, const float* __restrict__ Y,
                             float* nx, float* ny) {
  int idx = blockIdx.x * 256 + threadIdx.x;
  if (idx >= 2 * NPAD) return;
  int which = idx >= NPAD;
  int i = idx & (NPAD - 1);
  const float* S = which ? Y : X;
  float s = 1.0e30f;                    // sentinel: padded rows never win
  if (i < N1) {
    s = 0.0f;
    for (int c = 0; c < C_CH; ++c)
      #pragma unroll
      for (int p = 0; p < PATCH; ++p) {
        float v = S[c * T_LEN + i + p];
        s += v * v;
      }
  }
  (which ? ny : nx)[i] = s;
}

// --------------------------------------------------- WMMA frag loaders ----
union Frag16 { v16h v; unsigned u[8]; };

// A 16x32 f16 layout (ISA 7.12.2): lanes 0-15 rows M, K pairs {0..7,16..23};
// lanes 16-31 same rows, K pairs {8..15,24..31}.
__device__ inline v16h load_a_frag(const _Float16* s, int row, int halfId) {
  Frag16 f;
  #pragma unroll
  for (int v = 0; v < 8; ++v) {
    int k = 2 * v + (v >= 4 ? 8 : 0) + 8 * halfId;
    f.u[v] = *(const unsigned*)&s[row * LDS_PITCH + k];
  }
  return f.v;
}

// B 32x16 f16: lane = column N; lanes 0-15 hold K {0..7,16..23}, lanes 16-31 K {8..15,24..31}.
__device__ inline v16h load_b_frag(const _Float16* s, int col, int halfId) {
  Frag16 f;
  #pragma unroll
  for (int v = 0; v < 8; ++v) {
    int k = 2 * v + (v >= 4 ? 8 : 0) + 8 * halfId;
    f.u[v] = *(const unsigned*)&s[col * LDS_PITCH + k];
  }
  return f.v;
}

// --------------------------------------------------------- GEMM passes ----
// PASS 0: column-min of dist into colmin_bits (u32 float-bit atomicMin).
// PASS 1: row argmin of dist/(alpha+colmin) into best (u64 packed atomicMin).
template <int PASS>
__global__ __launch_bounds__(256)
void gemm_pass(const _Float16* __restrict__ XpH, const _Float16* __restrict__ XpL,
               const _Float16* __restrict__ YpH, const _Float16* __restrict__ YpL,
               const float* __restrict__ nx, const float* __restrict__ ny,
               const float* __restrict__ invc,
               unsigned* colmin_bits, unsigned long long* best) {
  const int tid   = threadIdx.x;
  const int i0    = blockIdx.y * 128;     // rows (x patches)
  const int j0    = blockIdx.x * 128;     // cols (y patches)
  const int wave  = tid >> 5;
  const int lane  = tid & 31;
  const int waveM = wave >> 1;            // 0..3 -> 32-row strips
  const int waveN = wave & 1;             // 0..1 -> 64-col strips
  const int n16    = lane & 15;
  const int halfId = lane >> 4;

  v8f acc[2][4];
  #pragma unroll
  for (int mf = 0; mf < 2; ++mf)
    #pragma unroll
    for (int nf = 0; nf < 4; ++nf)
      acc[mf][nf] = (v8f){0.f, 0.f, 0.f, 0.f, 0.f, 0.f, 0.f, 0.f};

#if USE_ASYNC_LDS
  // -------- double-buffered async staging: stage[2] x {Ah,Al,Bh,Bl} -------
  __shared__ __align__(16) _Float16 sbuf[2][4][128 * LDS_PITCH];

  auto issue = [&](int buf, int kb) {
    #pragma unroll
    for (int it = 0; it < 2; ++it) {      // 512 chunks/matrix, 256 lanes, 2 iters
      int cidx = it * 256 + tid;
      int r = cidx >> 2, q = cidx & 3;
      int go = kb + q * 8;
      int lo = r * LDS_PITCH + q * 8;
      async_cp16(&XpH[(size_t)(i0 + r) * KPAD + go], &sbuf[buf][0][lo]);
      async_cp16(&XpL[(size_t)(i0 + r) * KPAD + go], &sbuf[buf][1][lo]);
      async_cp16(&YpH[(size_t)(j0 + r) * KPAD + go], &sbuf[buf][2][lo]);
      async_cp16(&YpL[(size_t)(j0 + r) * KPAD + go], &sbuf[buf][3][lo]);
    }
  };

  issue(0, 0);
  for (int ks = 0; ks < KPAD / 32; ++ks) {
    __builtin_amdgcn_s_wait_asynccnt(0);  // this wave's stage-ks copies done
    __syncthreads();                      // all waves' copies visible; prev compute done
    if (ks + 1 < KPAD / 32) issue((ks + 1) & 1, (ks + 1) * 32);
    const _Float16* cAh = sbuf[ks & 1][0];
    const _Float16* cAl = sbuf[ks & 1][1];
    const _Float16* cBh = sbuf[ks & 1][2];
    const _Float16* cBl = sbuf[ks & 1][3];
#else
  // ---------------- fallback: sync staging, single buffer -----------------
  __shared__ __align__(16) _Float16 sbuf[4][128 * LDS_PITCH];
  for (int ks = 0; ks < KPAD / 32; ++ks) {
    int kb = ks * 32;
    __syncthreads();
    #pragma unroll
    for (int it = 0; it < 2; ++it) {
      int lin = tid + it * 256;
      int r = lin >> 2, q = lin & 3;
      int dst = r * LDS_PITCH + q * 8;
      int srcA = (i0 + r) * KPAD + kb + q * 8;
      int srcB = (j0 + r) * KPAD + kb + q * 8;
      *(uint4*)&sbuf[0][dst] = *(const uint4*)&XpH[srcA];
      *(uint4*)&sbuf[1][dst] = *(const uint4*)&XpL[srcA];
      *(uint4*)&sbuf[2][dst] = *(const uint4*)&YpH[srcB];
      *(uint4*)&sbuf[3][dst] = *(const uint4*)&YpL[srcB];
    }
    __syncthreads();
    const _Float16* cAh = sbuf[0];
    const _Float16* cAl = sbuf[1];
    const _Float16* cBh = sbuf[2];
    const _Float16* cBl = sbuf[3];
#endif

    v16h aH[2], aL[2];
    #pragma unroll
    for (int mf = 0; mf < 2; ++mf) {
      int row = waveM * 32 + mf * 16 + n16;
      aH[mf] = load_a_frag(cAh, row, halfId);
      aL[mf] = load_a_frag(cAl, row, halfId);
    }
    #pragma unroll
    for (int nf = 0; nf < 4; ++nf) {
      int col = waveN * 64 + nf * 16 + n16;
      v16h bH = load_b_frag(cBh, col, halfId);
      v16h bL = load_b_frag(cBl, col, halfId);
      #pragma unroll
      for (int mf = 0; mf < 2; ++mf) {
        acc[mf][nf] = __builtin_amdgcn_wmma_f32_16x16x32_f16(
            false, aH[mf], false, bH, (short)0, acc[mf][nf], false, false);
        acc[mf][nf] = __builtin_amdgcn_wmma_f32_16x16x32_f16(
            false, aH[mf], false, bL, (short)0, acc[mf][nf], false, false);
        acc[mf][nf] = __builtin_amdgcn_wmma_f32_16x16x32_f16(
            false, aL[mf], false, bH, (short)0, acc[mf][nf], false, false);
      }
    }
  }

  // C/D layout: VGPR r -> M = r + 8*halfId, N = n16.
  if (PASS == 0) {
    #pragma unroll
    for (int nf = 0; nf < 4; ++nf) {
      int j = j0 + waveN * 64 + nf * 16 + n16;
      float nyj = ny[j];
      float dmin = 3.0e38f;
      #pragma unroll
      for (int mf = 0; mf < 2; ++mf)
        #pragma unroll
        for (int r = 0; r < 8; ++r) {
          int i = i0 + waveM * 32 + mf * 16 + halfId * 8 + r;
          float d = (nx[i] + nyj - 2.0f * acc[mf][nf][r]) * INV_D;  // nx sentinel kills pads
          dmin = fminf(dmin, d);
        }
      dmin = fminf(dmin, __shfl_xor(dmin, 16, 32));   // both lane halves share j
      if (halfId == 0 && j < N1)
        atomicMin(&colmin_bits[j], __float_as_uint(fmaxf(dmin, 0.0f)));
    }
  } else {
    float nyv[4], ivv[4]; int jv[4];
    #pragma unroll
    for (int nf = 0; nf < 4; ++nf) {
      int j = j0 + waveN * 64 + nf * 16 + n16;
      jv[nf] = j; nyv[nf] = ny[j]; ivv[nf] = invc[j];
    }
    #pragma unroll
    for (int mf = 0; mf < 2; ++mf)
      #pragma unroll
      for (int r = 0; r < 8; ++r) {
        int i = i0 + waveM * 32 + mf * 16 + halfId * 8 + r;
        unsigned long long key = ~0ull;
        #pragma unroll
        for (int nf = 0; nf < 4; ++nf) {
          float d  = (nx[i] + nyv[nf] - 2.0f * acc[mf][nf][r]) * INV_D;
          float nd = fmaxf(d * ivv[nf], 0.0f);
          unsigned long long k =
              ((unsigned long long)__float_as_uint(nd) << 32) | (unsigned)jv[nf];
          if (jv[nf] < N1) key = (k < key) ? k : key;
        }
        #pragma unroll
        for (int s = 1; s < 16; s <<= 1) {           // butterfly within lane half
          unsigned long long o = __shfl_xor(key, s, 32);
          key = (o < key) ? o : key;
        }
        if (n16 == 0 && i < N1) atomicMin(&best[i], key);
      }
  }
}

// --------------------------------------------- 1/(alpha + colmin) ---------
__global__ void inv_kernel(const unsigned* colmin_bits, float* invc) {
  int j = blockIdx.x * 256 + threadIdx.x;
  if (j < NPAD) invc[j] = 1.0f / (ALPHA_C + __uint_as_float(colmin_bits[j]));
}

// ------------------- nnf extraction + exact dist at match -----------------
__global__ void finalize_kernel(const float* __restrict__ X, const float* __restrict__ Y,
                                const float* __restrict__ nx, const float* __restrict__ ny,
                                const unsigned long long* __restrict__ best,
                                int* nnf, float* dists) {
  int i = blockIdx.x * 256 + threadIdx.x;
  if (i >= N1) return;
  int j = (int)(unsigned)(best[i] & 0xFFFFFFFFull);
  nnf[i] = j;
  float dot = 0.0f;
  for (int c = 0; c < C_CH; ++c)
    #pragma unroll
    for (int p = 0; p < PATCH; ++p)
      dot += X[c * T_LEN + i + p] * Y[c * T_LEN + j + p];
  dists[i] = (nx[i] + ny[j] - 2.0f * dot) * INV_D;
}

// --------------------------- overlap-add fold -----------------------------
__global__ void fold_kernel(const float* __restrict__ Y, const int* __restrict__ nnf,
                            float* __restrict__ out) {
  int idx = blockIdx.x * 256 + threadIdx.x;
  if (idx >= C_CH * T_LEN) return;
  int c = idx / T_LEN, t = idx - c * T_LEN;
  float acc = 0.0f; int cnt = 0;
  #pragma unroll
  for (int p = 0; p < PATCH; ++p) {
    int i = t - p;
    if (i >= 0 && i < N1) { acc += Y[c * T_LEN + nnf[i] + p]; ++cnt; }
  }
  out[idx] = acc / (float)cnt;
}

// ------------------------------- mean -------------------------------------
__global__ void mean_kernel(const float* __restrict__ dists, float* out) {
  __shared__ float red[1024];
  float s = 0.0f;
  for (int i = threadIdx.x; i < N1; i += 1024) s += dists[i];
  red[threadIdx.x] = s;
  __syncthreads();
  for (int w = 512; w > 0; w >>= 1) {
    if (threadIdx.x < w) red[threadIdx.x] += red[threadIdx.x + w];
    __syncthreads();
  }
  if (threadIdx.x == 0) out[0] = red[0] / (float)N1;
}

// ---------------------------------------------------------------------------
extern "C" void kernel_launch(void* const* d_in, const int* in_sizes, int n_in,
                              void* d_out, int out_size, void* d_ws, size_t ws_size,
                              hipStream_t stream) {
  const float* X = (const float*)d_in[0];
  const float* Y = (const float*)d_in[1];
  float* out = (float*)d_out;                 // [72*8192 blended][1 mean]

  size_t off = 0;
  char* base = (char*)d_ws;
  auto alloc = [&](size_t bytes) -> void* {
    void* p = base + off;
    off += (bytes + 255) & ~(size_t)255;
    return p;
  };
  _Float16* XpH = (_Float16*)alloc((size_t)NPAD * KPAD * 2);
  _Float16* XpL = (_Float16*)alloc((size_t)NPAD * KPAD * 2);
  _Float16* YpH = (_Float16*)alloc((size_t)NPAD * KPAD * 2);
  _Float16* YpL = (_Float16*)alloc((size_t)NPAD * KPAD * 2);
  float*    nx  = (float*)alloc(NPAD * 4);
  float*    ny  = (float*)alloc(NPAD * 4);
  float*    invc = (float*)alloc(NPAD * 4);
  unsigned* colmin_bits = (unsigned*)alloc(NPAD * 4);
  unsigned long long* best = (unsigned long long*)alloc(NPAD * 8);
  int*      nnf = (int*)alloc(NPAD * 4);
  float*    dists = (float*)alloc(NPAD * 4);

  init_kernel<<<NPAD / 256, 256, 0, stream>>>(colmin_bits, best);
  prep_patches<<<(NPAD * KPAD) / 256, 256, 0, stream>>>(X, Y, XpH, XpL, YpH, YpL);
  norms_kernel<<<(2 * NPAD) / 256, 256, 0, stream>>>(X, Y, nx, ny);

  dim3 grid(NPAD / 128, NPAD / 128);
  gemm_pass<0><<<grid, 256, 0, stream>>>(XpH, XpL, YpH, YpL, nx, ny, invc,
                                         colmin_bits, best);
  inv_kernel<<<NPAD / 256, 256, 0, stream>>>(colmin_bits, invc);
  gemm_pass<1><<<grid, 256, 0, stream>>>(XpH, XpL, YpH, YpL, nx, ny, invc,
                                         colmin_bits, best);

  finalize_kernel<<<(N1 + 255) / 256, 256, 0, stream>>>(X, Y, nx, ny, best, nnf, dists);
  fold_kernel<<<(C_CH * T_LEN) / 256, 256, 0, stream>>>(Y, nnf, out);
  mean_kernel<<<1, 1024, 0, stream>>>(dists, out + C_CH * T_LEN);
}
